// Model_73315091743525
// MI455X (gfx1250) — matrix operand order repaired and checked by reading, not compiled, not run on hardware
//
#include <hip/hip_runtime.h>
#include <hip/hip_bf16.h>
#include <math.h>

#define B_   256
#define T_   1024
#define IN_  128
#define K_   256
#define M_   127
#define BT   16              // batch rows per workgroup (WMMA M-tile)
#define XH_PITCH 392         // halves per row (mult of 8, padded vs 384)

#define WCAT_JT 16           // 256/16 hidden column tiles
#define WCAT_KT 12           // 384/32 K-steps
#define WCAT_FRAGS (WCAT_JT*WCAT_KT)   // 192
#define WV_MT 8              // 128/16 logit column tiles
#define WV_KT 8              // 256/32 K-steps
#define WV_FRAGS (WV_MT*WV_KT)         // 64
#define XKT 4                // K-steps covering the x part (128/32)
#define HKT 8                // K-steps covering the h part (256/32)

typedef __attribute__((ext_vector_type(16))) __bf16 bf16x16;
typedef __attribute__((ext_vector_type(8)))  __bf16 bf16x8;
typedef __attribute__((ext_vector_type(8)))  float  f32x8;

__device__ __forceinline__ float fast_tanh(float v) {
#if __has_builtin(__builtin_amdgcn_tanhf)
    return __builtin_amdgcn_tanhf(v);   // v_tanh_f32 (gfx1250 TRANS op)
#else
    return tanhf(v);
#endif
}

__global__ __launch_bounds__(256, 1)
void rnn_fused_kernel(const float* __restrict__ x,
                      const float* __restrict__ Wih,
                      const float* __restrict__ Whh,
                      const float* __restrict__ bih,
                      const float* __restrict__ bhh,
                      const float* __restrict__ Wv,
                      const float* __restrict__ bv,
                      float* __restrict__ out)
{
    // LDS: 192KB + 64KB + 12.25KB + 1.5KB  ~= 270KB of the WGP's 320KB
    __shared__ __align__(32) __bf16 sWcat[WCAT_FRAGS * 512]; // packed B-fragments of [W_ih;W_hh]^T
    __shared__ __align__(32) __bf16 sWv  [WV_FRAGS   * 512]; // packed B-fragments of W_v^T (padded to 128 cols)
    __shared__ __align__(32) __bf16 sXH  [BT * XH_PITCH];    // live [x_t | h_t] tile, row-major
    __shared__ float sBsum[K_];
    __shared__ float sBv[128];

    const int tid  = threadIdx.x;
    const int lane = tid & 31;
    const int wave = tid >> 5;
    const int b0   = blockIdx.x * BT;

    // ---------------- one-time: pack weights into WMMA B-fragment layout ----------------
    // B (32x16) per-lane layout used here: col = lane%16; lanes 0-15 hold K 0..15, lanes 16-31 hold K 16..31,
    // stored contiguously 16 halves per lane -> reads are 2x ds_load_b128.
    for (int idx = tid; idx < WCAT_FRAGS * 512; idx += 256) {
        int f  = idx >> 9;          // fragment
        int l  = (idx >> 4) & 31;   // lane slot
        int e  = idx & 15;          // element
        int kt = f % WCAT_KT, jt = f / WCAT_KT;
        int k  = kt * 32 + ((l & 16) ? 16 : 0) + e;
        int j  = jt * 16 + (l & 15);
        float v = (k < IN_) ? Wih[j * IN_ + k] : Whh[j * K_ + (k - IN_)];
        sWcat[idx] = (__bf16)v;
    }
    for (int idx = tid; idx < WV_FRAGS * 512; idx += 256) {
        int f  = idx >> 9;
        int l  = (idx >> 4) & 31;
        int e  = idx & 15;
        int kt = f % WV_KT, mt = f / WV_KT;
        int k  = kt * 32 + ((l & 16) ? 16 : 0) + e;
        int m  = mt * 16 + (l & 15);
        float v = (m < M_) ? Wv[m * K_ + k] : 0.0f;
        sWv[idx] = (__bf16)v;
    }
    if (tid < K_)  sBsum[tid] = bih[tid] + bhh[tid];
    if (tid < 128) sBv[tid]   = (tid < M_) ? bv[tid] : 0.0f;
    // zero the h region (+ pad columns); x region is filled below (disjoint)
    for (int idx = tid; idx < BT * (XH_PITCH - IN_); idx += 256) {
        int row = idx / (XH_PITCH - IN_);
        int c   = IN_ + idx % (XH_PITCH - IN_);
        sXH[row * XH_PITCH + c] = (__bf16)0.0f;
    }

    // x tile loader: 16 rows x 128 f32 -> bf16 into sXH[:,0:128]; thread = (row, 8-elem segment)
    auto load_x = [&](int t) {
        int row = tid >> 4, seg = tid & 15;
        const float* xp = &x[((size_t)(b0 + row) * T_ + t) * IN_ + seg * 8];
        float4 v0 = ((const float4*)xp)[0];
        float4 v1 = ((const float4*)xp)[1];
        bf16x8 p;
        p[0] = (__bf16)v0.x; p[1] = (__bf16)v0.y; p[2] = (__bf16)v0.z; p[3] = (__bf16)v0.w;
        p[4] = (__bf16)v1.x; p[5] = (__bf16)v1.y; p[6] = (__bf16)v1.z; p[7] = (__bf16)v1.w;
        *(bf16x8*)&sXH[row * XH_PITCH + seg * 8] = p;
        if (t + 2 < T_) __builtin_prefetch(xp + 2 * IN_, 0, 0);  // global_prefetch_b8
    };

    // A fragment (16x32 bf16) from the row-major [x|h] tile.
    // Lane l: row = l%16; halves 0-15 = K[kb+off .. +15], off = (l>=16)?16:0
    auto load_a = [&](int kb) -> bf16x16 {
        int row = lane & 15;
        int off = (lane & 16) ? 16 : 0;
        const __bf16* p = &sXH[row * XH_PITCH + kb + off];
        bf16x8 lo = *(const bf16x8*)p;
        bf16x8 hi = *(const bf16x8*)(p + 8);
        bf16x16 a;
#pragma unroll
        for (int i = 0; i < 8; ++i) { a[i] = lo[i]; a[i + 8] = hi[i]; }
        return a;
    };
    auto load_b = [&](const __bf16* base, int f) -> bf16x16 {
        return *(const bf16x16*)&base[(f * 32 + lane) * 16];
    };

    load_x(0);
    __syncthreads();

    const int jt0  = wave * 2;            // 2 hidden column tiles per wave
    const int mt   = wave;                // 1 logit column tile per wave
    const int wrow = (lane & 16) ? 8 : 0; // C/D layout row offset
    const int cn   = lane & 15;           // C/D layout column within tile

    // Hoist ALL loop-invariant weight B-fragments into registers (gfx1250 has a
    // 1024-VGPR file; the compiler reaches the high range via s_set_vgpr_msb).
    // 24 Wcat fragments (192 VGPRs) + 8 Wv fragments (64 VGPRs), reused 1024x.
    bf16x16 wc0[WCAT_KT], wc1[WCAT_KT];
#pragma unroll
    for (int kt = 0; kt < WCAT_KT; ++kt) {
        wc0[kt] = load_b(sWcat, jt0 * WCAT_KT + kt);
        wc1[kt] = load_b(sWcat, (jt0 + 1) * WCAT_KT + kt);
    }
    bf16x16 wvreg[WV_KT];
#pragma unroll
    for (int kt = 0; kt < WV_KT; ++kt) wvreg[kt] = load_b(sWv, mt * WV_KT + kt);

    const float bs0 = sBsum[jt0 * 16 + cn];
    const float bs1 = sBsum[(jt0 + 1) * 16 + cn];
    const float bvv = sBv[mt * 16 + cn];
    const int   m   = mt * 16 + cn;

    // h-part A-fragments live in registers across iterations: they serve both the
    // logit matmul of step t AND the recurrent matmul of step t+1 (same bytes).
    bf16x16 ha[HKT];
#pragma unroll
    for (int kt = 0; kt < HKT; ++kt) ha[kt] = load_a(IN_ + kt * 32);  // h_{-1} = 0

    for (int t = 0; t < T_; ++t) {
        // -------- recurrent matmul: [16,384] @ Wcat -> pre-activation for 2 j-tiles --------
        f32x8 acc0, acc1;
#pragma unroll
        for (int r = 0; r < 8; ++r) { acc0[r] = bs0; acc1[r] = bs1; }

#pragma unroll
        for (int kt = 0; kt < XKT; ++kt) {   // x part: fresh A-fragments from LDS
            bf16x16 a = load_a(kt * 32);
            acc0 = __builtin_amdgcn_wmma_f32_16x16x32_bf16(false, a, false, wc0[kt], (short)0, acc0, false, false);
            acc1 = __builtin_amdgcn_wmma_f32_16x16x32_bf16(false, a, false, wc1[kt], (short)0, acc1, false, false);
        }
#pragma unroll
        for (int kt = 0; kt < HKT; ++kt) {   // h part: A-fragments carried in registers
            acc0 = __builtin_amdgcn_wmma_f32_16x16x32_bf16(false, ha[kt], false, wc0[XKT + kt], (short)0, acc0, false, false);
            acc1 = __builtin_amdgcn_wmma_f32_16x16x32_bf16(false, ha[kt], false, wc1[XKT + kt], (short)0, acc1, false, false);
        }
        __bf16 h0[8], h1[8];
#pragma unroll
        for (int r = 0; r < 8; ++r) {
            h0[r] = (__bf16)fast_tanh(acc0[r]);
            h1[r] = (__bf16)fast_tanh(acc1[r]);
        }

        __syncthreads();   // everyone done reading the x region of sXH

        // write h_t into the [x|h] tile (C/D layout scatter) and stage x_{t+1}
#pragma unroll
        for (int r = 0; r < 8; ++r) {
            sXH[(r + wrow) * XH_PITCH + IN_ + jt0 * 16 + cn]       = h0[r];
            sXH[(r + wrow) * XH_PITCH + IN_ + (jt0 + 1) * 16 + cn] = h1[r];
        }
        if (t + 1 < T_) load_x(t + 1);

        __syncthreads();   // h_t (and x_{t+1}) visible

        // refresh the h A-fragments once; reused by logits now and recurrence at t+1
#pragma unroll
        for (int kt = 0; kt < HKT; ++kt) ha[kt] = load_a(IN_ + kt * 32);

        // -------- output projection: [16,256] @ Wv^T -> one 16-col logit tile --------
        f32x8 accv;
#pragma unroll
        for (int r = 0; r < 8; ++r) accv[r] = bvv;
#pragma unroll
        for (int kt = 0; kt < WV_KT; ++kt) {
            accv = __builtin_amdgcn_wmma_f32_16x16x32_bf16(false, ha[kt], false, wvreg[kt], (short)0, accv, false, false);
        }
        if (m < M_) {
#pragma unroll
            for (int r = 0; r < 8; ++r) {
                out[((size_t)(b0 + r + wrow) * T_ + t) * M_ + m] = accv[r];
            }
        }
        // next iteration reads only the x region of sXH (h comes from ha[])
    }
}

extern "C" void kernel_launch(void* const* d_in, const int* in_sizes, int n_in,
                              void* d_out, int out_size, void* d_ws, size_t ws_size,
                              hipStream_t stream) {
    (void)in_sizes; (void)n_in; (void)out_size; (void)d_ws; (void)ws_size;
    const float* x   = (const float*)d_in[0];
    const float* Wih = (const float*)d_in[1];
    const float* Whh = (const float*)d_in[2];
    const float* bih = (const float*)d_in[3];
    const float* bhh = (const float*)d_in[4];
    const float* Wv  = (const float*)d_in[5];
    const float* bv  = (const float*)d_in[6];
    rnn_fused_kernel<<<B_ / BT, 256, 0, stream>>>(x, Wih, Whh, bih, bhh, Wv, bv, (float*)d_out);
}
